// KettleGraphReasoner_27273042329632
// MI455X (gfx1250) — matrix operand order repaired and checked by reading, not compiled, not run on hardware
//
#include <hip/hip_runtime.h>
#include <hip/hip_bf16.h>
#include <math.h>

// ---------------------------------------------------------------------------
// KettleGraphReasoner on MI455X (gfx1250, wave32, WMMA)
//
// Dense GEMMs (N x K x 64) via v_wmma_f32_16x16x32_f16 (f16 in, f32 accum),
// with the A tile staged global->LDS once per block via the CDNA5 async
// copy path (global_load_async_to_lds_b128 + s_wait_asynccnt).
// Edge work factored into per-node / per-type precomputes + gather/pointwise.
// Softmax via order-preserving uint atomicMax + float atomicAdd.
// ---------------------------------------------------------------------------

typedef __attribute__((ext_vector_type(16))) _Float16 v16h;
typedef __attribute__((ext_vector_type(8)))  float    v8f;

#define NN 50000            // N_NODES
#define NE 800000           // N_EDGES
#define NH (NN * 64)        // node feature plane (H = 64)
#define NFDIM 256
#define HDIM 64
#define TDDIM 32
#define NTYPES 64
#define QDIM 128
#define NLAYERS 3

// ---------------------------------------------------------------------------
// small helpers
// ---------------------------------------------------------------------------
__device__ __forceinline__ unsigned f2ord(float f) {
    unsigned u = __float_as_uint(f);
    return (u & 0x80000000u) ? ~u : (u | 0x80000000u);
}
__device__ __forceinline__ float ord2f(unsigned k) {
    unsigned u = (k & 0x80000000u) ? (k & 0x7fffffffu) : ~k;
    return __uint_as_float(u);
}
__device__ __forceinline__ float wave_sum(float v) {
#pragma unroll
    for (int o = 16; o > 0; o >>= 1) v += __shfl_xor(v, o, 32);
    return v;
}
__device__ __forceinline__ float sigmoidf(float x) { return 1.0f / (1.0f + expf(-x)); }

// ---------------------------------------------------------------------------
// zero fill (used for m/denom/agg init)
// ---------------------------------------------------------------------------
__global__ void k_fill_zero(unsigned* p, int n) {
    int i = blockIdx.x * blockDim.x + threadIdx.x;
    if (i < n) p[i] = 0u;
}

// ---------------------------------------------------------------------------
// type_emb = tanh(edge_descriptor @ Ws + bs)   (64 x 32), also to d_out tail
// ---------------------------------------------------------------------------
__global__ void k_type_emb(const float* __restrict__ ed, const float* __restrict__ Ws,
                           const float* __restrict__ bs, float* __restrict__ te,
                           float* __restrict__ out_te) {
    int i = blockIdx.x * blockDim.x + threadIdx.x;
    if (i >= NTYPES * TDDIM) return;
    int t = i >> 5, d = i & 31;
    float acc = bs[d];
#pragma unroll 4
    for (int e = 0; e < 64; ++e) acc += ed[t * 64 + e] * Ws[e * TDDIM + d];
    float v = tanhf(acc);
    te[i] = v;
    out_te[i] = v;
}

// q = query @ Wq + bq  (64)
__global__ void k_qproj(const float* __restrict__ query, const float* __restrict__ Wq,
                        const float* __restrict__ bq, float* __restrict__ q) {
    int j = threadIdx.x;
    if (j >= HDIM) return;
    float acc = bq[j];
#pragma unroll 4
    for (int i = 0; i < QDIM; ++i) acc += query[i] * Wq[i * HDIM + j];
    q[j] = acc;
}

// qn = q @ Wns1[64:128] + bns1 ; qe = q @ Wes1[160:224] + bes1
__global__ void k_qn_qe(const float* __restrict__ q, const float* __restrict__ Wns1,
                        const float* __restrict__ bns1, const float* __restrict__ Wes1,
                        const float* __restrict__ bes1, float* __restrict__ qn,
                        float* __restrict__ qe) {
    int idx = threadIdx.x;
    if (idx >= 128) return;
    int j = idx & 63;
    if (idx < 64) {
        float a = bns1[j];
#pragma unroll 4
        for (int i = 0; i < 64; ++i) a += q[i] * Wns1[(64 + i) * HDIM + j];
        qn[j] = a;
    } else {
        float a = bes1[j];
#pragma unroll 4
        for (int i = 0; i < 64; ++i) a += q[i] * Wes1[(160 + i) * HDIM + j];
        qe[j] = a;
    }
}

// P3[t][j] = type_emb[t] @ Wes1[128:160]   (64 x 64)
__global__ void k_p3(const float* __restrict__ te, const float* __restrict__ Wes1,
                     float* __restrict__ P3) {
    int i = blockIdx.x * blockDim.x + threadIdx.x;
    if (i >= NTYPES * HDIM) return;
    int t = i >> 6, j = i & 63;
    float a = 0.0f;
#pragma unroll 4
    for (int d = 0; d < TDDIM; ++d) a += te[t * TDDIM + d] * Wes1[(128 + d) * HDIM + j];
    P3[i] = a;
}

// dte[l][t] = type_emb[t] . attn_a[l][128:160]
__global__ void k_dte(const float* __restrict__ te, const float* __restrict__ attn_a,
                      float* __restrict__ dte) {
    int i = threadIdx.x;
    if (i >= NLAYERS * NTYPES) return;
    int l = i >> 6, t = i & 63;
    float a = 0.0f;
#pragma unroll 4
    for (int d = 0; d < TDDIM; ++d) a += te[t * TDDIM + d] * attn_a[l * 160 + 128 + d];
    dte[i] = a;
}

// ---------------------------------------------------------------------------
// Pack a (K x 64) fp32 weight (row stride 64, starting at row0) into the
// per-lane WMMA B fragment layout (16x16x32, f16):
//   element idx = ((kt*4 + ntile)*32 + lane)*16 + i
//   lanes 0-15 : col = ntile*16 + lane,     K = kt*32 + {0..7,16..23}[i]
//   lanes16-31 : col = ntile*16 + lane-16,  K = kt*32 + {8..15,24..31}[i]
// ---------------------------------------------------------------------------
__global__ void k_pack_w(const float* __restrict__ W, int row0, int K,
                         _Float16* __restrict__ out) {
    int idx = blockIdx.x * blockDim.x + threadIdx.x;
    if (idx >= K * 64) return;
    int i    = idx & 15;
    int lane = (idx >> 4) & 31;
    int nt   = (idx >> 9) & 3;
    int kt   = idx >> 11;
    int k    = kt * 32 + ((lane >> 4) << 3) + ((i >> 3) << 4) + (i & 7);
    int col  = (nt << 4) + (lane & 15);
    out[idx] = (_Float16)W[(row0 + k) * 64 + col];
}

// ---------------------------------------------------------------------------
// WMMA GEMM: C(50000 x 64) = A(50000 x K) @ Bpacked + bias, then * scale.
// Block = 128 threads = 4 waves; wave w owns N-tile w; blockIdx.x = M-tile.
// The 16x32 fp32 A tile is staged global->LDS once per block per k-step via
// CDNA5 async copy (128 lanes x 16B = 2KB), so the 4 waves share it instead
// of re-loading it 4x from global. Each wave then builds its f16 A fragment
// (lanes 0-15: K {0..7,16..23}; lanes 16-31: K {8..15,24..31}) from LDS.
// ---------------------------------------------------------------------------
__global__ void __launch_bounds__(128) k_gemm(const float* __restrict__ A, int lda, int ktiles,
                                              const _Float16* __restrict__ pB,
                                              const float* __restrict__ bias,
                                              const float* __restrict__ scale_ptr,
                                              float* __restrict__ C) {
    __shared__ float sA[16 * 32];   // one 16x32 fp32 A tile (2 KB)

    int tid   = threadIdx.x;
    int mtile = blockIdx.x;
    int w     = tid >> 5;
    int lane  = tid & 31;
    int hi    = lane >> 4;

    // this thread's slice of the cooperative async copy: 16B per lane
    int crow   = tid >> 3;        // 0..15
    int cchunk = (tid & 7) << 2;  // 0,4,...,28 (floats)
    unsigned lds_dst = (unsigned)(size_t)&sA[crow * 32 + cchunk];

    v8f acc = {};
    for (int kt = 0; kt < ktiles; ++kt) {
        // async global->LDS stage of A[mtile*16 .. +16, kt*32 .. +32)
        const float* gp = A + (mtile * 16 + crow) * lda + kt * 32 + cchunk;
        asm volatile("global_load_async_to_lds_b128 %0, %1, off"
                     :: "v"(lds_dst), "v"((unsigned long long)(size_t)gp)
                     : "memory");
        asm volatile("s_wait_asynccnt 0x0" ::: "memory");
        __syncthreads();

        // build f16 A fragment from the shared tile
        const float* ap = &sA[(lane & 15) * 32 + hi * 8];
        union { v16h v; _Float16 e[16]; } af;
#pragma unroll
        for (int i = 0; i < 8; ++i) af.e[i] = (_Float16)ap[i];
#pragma unroll
        for (int i = 0; i < 8; ++i) af.e[8 + i] = (_Float16)ap[16 + i];

        v16h b = *(const v16h*)(pB + ((((kt << 2) + w) * 32 + lane) << 4));
        acc = __builtin_amdgcn_wmma_f32_16x16x32_f16(false, af.v, false, b,
                                                     (short)0, acc, false, false);
        __syncthreads();   // tile consumed before next iteration overwrites it
    }
    float sc  = scale_ptr ? scale_ptr[0] : 1.0f;
    int   col = (w << 4) + (lane & 15);
    float bv  = bias ? bias[col] : 0.0f;
    int   rb  = mtile * 16 + hi * 8;
    union { v8f v; float e[8]; } df;
    df.v = acc;
#pragma unroll
    for (int j = 0; j < 8; ++j)
        C[(rb + j) * HDIM + col] = (df.e[j] + bv) * sc;
}

// ---------------------------------------------------------------------------
// expmap0 / logmap0 : one wave per node, 2 features per lane (H = 64)
// ---------------------------------------------------------------------------
__global__ void k_expmap0(const float* __restrict__ X, float* __restrict__ H, int do_relu) {
    int node = blockIdx.x * 8 + (threadIdx.x >> 5);
    if (node >= NN) return;
    int lane = threadIdx.x & 31;
    float x0 = X[node * HDIM + lane];
    float x1 = X[node * HDIM + lane + 32];
    if (do_relu) { x0 = fmaxf(x0, 0.0f); x1 = fmaxf(x1, 0.0f); }
    float ss = wave_sum(x0 * x0 + x1 * x1);
    float n  = fmaxf(sqrtf(ss), 1e-15f);
    float sc = tanhf(n) / n;
    H[node * HDIM + lane]      = x0 * sc;
    H[node * HDIM + lane + 32] = x1 * sc;
}

__global__ void k_logmap0(const float* __restrict__ X, float* __restrict__ O,
                          float* __restrict__ copy_out) {
    int node = blockIdx.x * 8 + (threadIdx.x >> 5);
    if (node >= NN) return;
    int lane = threadIdx.x & 31;
    float x0 = X[node * HDIM + lane];
    float x1 = X[node * HDIM + lane + 32];
    float ss = wave_sum(x0 * x0 + x1 * x1);
    float n  = fmaxf(sqrtf(ss), 1e-15f);
    float p  = fminf(n, 1.0f - 1e-5f);
    float sc = atanhf(p) / n;
    O[node * HDIM + lane]      = x0 * sc;
    O[node * HDIM + lane + 32] = x1 * sc;
    if (copy_out) {
        copy_out[node * HDIM + lane]      = x0;
        copy_out[node * HDIM + lane + 32] = x1;
    }
}

// per-node attention dots: dsrc = ht . a[0:64], ddst = ht . a[64:128]
__global__ void k_attn_dots(const float* __restrict__ ht, const float* __restrict__ a,
                            float* __restrict__ dsrc, float* __restrict__ ddst) {
    int node = blockIdx.x * 8 + (threadIdx.x >> 5);
    if (node >= NN) return;
    int lane = threadIdx.x & 31;
    float h0 = ht[node * HDIM + lane];
    float h1 = ht[node * HDIM + lane + 32];
    float s  = wave_sum(h0 * a[lane] + h1 * a[lane + 32]);
    float d  = wave_sum(h0 * a[64 + lane] + h1 * a[96 + lane]);
    if (lane == 0) { dsrc[node] = s; ddst[node] = d; }
}

// edge attention logit + segment max (order-preserving uint atomicMax)
__global__ void k_edge_s_max(const int* __restrict__ src, const int* __restrict__ dst,
                             const int* __restrict__ et, const float* __restrict__ dsrc,
                             const float* __restrict__ ddst, const float* __restrict__ dte_l,
                             float* __restrict__ s_buf, unsigned* __restrict__ m_u) {
    int e = blockIdx.x * blockDim.x + threadIdx.x;
    if (e >= NE) return;
    float s = dsrc[src[e]] + ddst[dst[e]] + dte_l[et[e]];
    s = (s > 0.0f) ? s : 0.2f * s;          // leaky_relu(0.2)
    s_buf[e] = s;
    atomicMax(&m_u[dst[e]], f2ord(s));
}

// ex = exp(s - max[dst]); denom[dst] += ex
__global__ void k_edge_exp(const int* __restrict__ dst, const float* __restrict__ s_buf,
                           const unsigned* __restrict__ m_u, float* __restrict__ ex_buf,
                           float* __restrict__ den) {
    int e = blockIdx.x * blockDim.x + threadIdx.x;
    if (e >= NE) return;
    int d = dst[e];
    float ex = expf(s_buf[e] - ord2f(m_u[d]));
    ex_buf[e] = ex;
    atomicAdd(&den[d], ex);
}

// agg[dst] += alpha * msg[src]   (one wave per edge, 2 features/lane)
__global__ void k_scatter(const int* __restrict__ src, const int* __restrict__ dst,
                          const float* __restrict__ ex_buf, const float* __restrict__ den,
                          const float* __restrict__ msg, float* __restrict__ agg) {
    int e = blockIdx.x * 8 + (threadIdx.x >> 5);
    if (e >= NE) return;
    int lane = threadIdx.x & 31;
    int s = src[e], d = dst[e];
    float alpha = ex_buf[e] / (den[d] + 1e-15f);
    atomicAdd(&agg[d * HDIM + lane],      alpha * msg[s * HDIM + lane]);
    atomicAdd(&agg[d * HDIM + lane + 32], alpha * msg[s * HDIM + lane + 32]);
}

// node head: sigmoid(relu(Z + qn) . Wns2 + bns2)
__global__ void k_node_score(const float* __restrict__ Z, const float* __restrict__ qn,
                             const float* __restrict__ Wns2, const float* __restrict__ bns2,
                             float* __restrict__ out) {
    int node = blockIdx.x * 8 + (threadIdx.x >> 5);
    if (node >= NN) return;
    int lane = threadIdx.x & 31;
    float z0 = fmaxf(Z[node * HDIM + lane]      + qn[lane],      0.0f);
    float z1 = fmaxf(Z[node * HDIM + lane + 32] + qn[lane + 32], 0.0f);
    float acc = wave_sum(z0 * Wns2[lane] + z1 * Wns2[lane + 32]);
    if (lane == 0) out[node] = sigmoidf(acc + bns2[0]);
}

// edge head: sigmoid(relu(P1[s]+P2[d]+P3[t]+qe) . Wes2 + bes2)
__global__ void k_edge_score(const int* __restrict__ src, const int* __restrict__ dst,
                             const int* __restrict__ et, const float* __restrict__ P1,
                             const float* __restrict__ P2, const float* __restrict__ P3,
                             const float* __restrict__ qe, const float* __restrict__ Wes2,
                             const float* __restrict__ bes2, float* __restrict__ out) {
    int e = blockIdx.x * 8 + (threadIdx.x >> 5);
    if (e >= NE) return;
    int lane = threadIdx.x & 31;
    int s = src[e], d = dst[e], t = et[e];
    float acc = 0.0f;
#pragma unroll
    for (int k = 0; k < 2; ++k) {
        int j = lane + 32 * k;
        float z = P1[s * HDIM + j] + P2[d * HDIM + j] + P3[t * HDIM + j] + qe[j];
        acc += fmaxf(z, 0.0f) * Wes2[j];
    }
    acc = wave_sum(acc);
    if (lane == 0) out[e] = sigmoidf(acc + bes2[0]);
}

// ---------------------------------------------------------------------------
// orchestration
// ---------------------------------------------------------------------------
extern "C" void kernel_launch(void* const* d_in, const int* in_sizes, int n_in,
                              void* d_out, int out_size, void* d_ws, size_t ws_size,
                              hipStream_t stream) {
    (void)in_sizes; (void)n_in; (void)out_size; (void)ws_size;

    const float* node_features = (const float*)d_in[0];
    const float* edge_desc     = (const float*)d_in[1];
    const float* query         = (const float*)d_in[2];
    const float* Wn            = (const float*)d_in[3];
    const float* bn            = (const float*)d_in[4];
    const float* tangent_scale = (const float*)d_in[5];
    const float* Wq            = (const float*)d_in[6];
    const float* bq            = (const float*)d_in[7];
    const float* Ws            = (const float*)d_in[8];
    const float* bs            = (const float*)d_in[9];
    const float* attn_a        = (const float*)d_in[10];
    const float* Wmp           = (const float*)d_in[11];
    const float* bmp           = (const float*)d_in[12];
    const float* Wns1          = (const float*)d_in[13];
    const float* bns1          = (const float*)d_in[14];
    const float* Wns2          = (const float*)d_in[15];
    const float* bns2          = (const float*)d_in[16];
    const float* Wes1          = (const float*)d_in[17];
    const float* bes1          = (const float*)d_in[18];
    const float* Wes2          = (const float*)d_in[19];
    const float* bes2          = (const float*)d_in[20];
    const int*   edge_index    = (const int*)d_in[21];
    const int*   etype         = (const int*)d_in[22];
    const int*   src           = edge_index;
    const int*   dst           = edge_index + NE;

    float* out_node = (float*)d_out;
    float* out_edge = out_node + NN;
    float* out_h    = out_edge + NE;
    float* out_te   = out_h + NH;

    // workspace layout
    float* fw  = (float*)d_ws;
    float* h   = fw; fw += NH;      // Poincare-ball embeddings
    float* B1  = fw; fw += NH;      // ht / h_flat
    float* B2  = fw; fw += NH;      // msg / P1
    float* B3  = fw; fw += NH;      // agg / P2
    float* B4  = fw; fw += NH;      // gemm-lin / Znode
    float* s_buf  = fw; fw += NE;
    float* ex_buf = fw; fw += NE;
    unsigned* m_u = (unsigned*)fw; fw += NN;   // m_u and den adjacent (joint zero-fill)
    float* den  = fw; fw += NN;
    float* dsrc = fw; fw += NN;
    float* ddst = fw; fw += NN;
    float* te   = fw; fw += NTYPES * TDDIM;
    float* qv   = fw; fw += 64;
    float* qn   = fw; fw += 64;
    float* qe   = fw; fw += 64;
    float* P3   = fw; fw += NTYPES * HDIM;
    float* dte  = fw; fw += NLAYERS * NTYPES;
    _Float16* pWn    = (_Float16*)fw;       // 256*64 halves
    _Float16* pWmp   = pWn + NFDIM * 64;    // 3 * 64*64
    _Float16* pWns1a = pWmp + NLAYERS * 64 * 64;
    _Float16* pWes1a = pWns1a + 64 * 64;
    _Float16* pWes1b = pWes1a + 64 * 64;

    // --- tiny precomputes -------------------------------------------------
    k_type_emb<<<8, 256, 0, stream>>>(edge_desc, Ws, bs, te, out_te);
    k_qproj<<<1, 64, 0, stream>>>(query, Wq, bq, qv);
    k_qn_qe<<<1, 128, 0, stream>>>(qv, Wns1, bns1, Wes1, bes1, qn, qe);
    k_p3<<<16, 256, 0, stream>>>(te, Wes1, P3);
    k_dte<<<1, 192, 0, stream>>>(te, attn_a, dte);

    // --- pack weights into WMMA B-fragment layout -------------------------
    k_pack_w<<<64, 256, 0, stream>>>(Wn, 0, NFDIM, pWn);
    for (int l = 0; l < NLAYERS; ++l)
        k_pack_w<<<16, 256, 0, stream>>>(Wmp + l * 64 * 64, 0, 64, pWmp + l * 64 * 64);
    k_pack_w<<<16, 256, 0, stream>>>(Wns1, 0, 64, pWns1a);
    k_pack_w<<<16, 256, 0, stream>>>(Wes1, 0, 64, pWes1a);
    k_pack_w<<<16, 256, 0, stream>>>(Wes1, 64, 64, pWes1b);

    // --- input embedding: (X @ Wn + bn) * tangent_scale, then expmap0 -----
    k_gemm<<<3125, 128, 0, stream>>>(node_features, NFDIM, NFDIM / 32, pWn, bn,
                                     tangent_scale, B4);
    k_expmap0<<<6250, 256, 0, stream>>>(B4, h, 0);

    // --- message-passing layers ------------------------------------------
    for (int l = 0; l < NLAYERS; ++l) {
        k_logmap0<<<6250, 256, 0, stream>>>(h, B1, nullptr);
        k_attn_dots<<<6250, 256, 0, stream>>>(B1, attn_a + l * 160, dsrc, ddst);
        k_fill_zero<<<(2 * NN + 255) / 256, 256, 0, stream>>>(m_u, 2 * NN);
        k_edge_s_max<<<NE / 256, 256, 0, stream>>>(src, dst, etype, dsrc, ddst,
                                                   dte + l * NTYPES, s_buf, m_u);
        k_edge_exp<<<NE / 256, 256, 0, stream>>>(dst, s_buf, m_u, ex_buf, den);
        k_fill_zero<<<(NH + 255) / 256, 256, 0, stream>>>((unsigned*)B3, NH);
        k_gemm<<<3125, 128, 0, stream>>>(B1, HDIM, 2, pWmp + l * 64 * 64,
                                         bmp + l * HDIM, nullptr, B2);
        k_scatter<<<NE / 8, 256, 0, stream>>>(src, dst, ex_buf, den, B2, B3);
        k_expmap0<<<6250, 256, 0, stream>>>(B3, h, 1);
    }

    // --- readout ----------------------------------------------------------
    k_logmap0<<<6250, 256, 0, stream>>>(h, B1, out_h);   // also copies h to output
    k_gemm<<<3125, 128, 0, stream>>>(B1, HDIM, 2, pWns1a, nullptr, nullptr, B4); // Znode
    k_gemm<<<3125, 128, 0, stream>>>(B1, HDIM, 2, pWes1a, nullptr, nullptr, B2); // P1
    k_gemm<<<3125, 128, 0, stream>>>(B1, HDIM, 2, pWes1b, nullptr, nullptr, B3); // P2
    k_node_score<<<6250, 256, 0, stream>>>(B4, qn, Wns2, bns2, out_node);
    k_edge_score<<<NE / 8, 256, 0, stream>>>(src, dst, etype, B2, B3, P3, qe,
                                             Wes2, bes2, out_edge);
}
